// MemoryMoCo_id_3083786519082
// MI455X (gfx1250) — compile-verified, matrix operand order checked by reading, not run on hardware
//
#include <hip/hip_runtime.h>
#include <cstdint>
#include <cstddef>

// ---------------------------------------------------------------------------
// MemoryMoCo circle-loss, fused for MI455X (gfx1250, wave32, WMMA, TDM).
// B=256, D=512, K=65536. Memory-bound: stream `memory` (128 MiB) exactly once,
// fusing GEMM (bf16 WMMA), queue writeback, and streaming logsumexp partials.
// ---------------------------------------------------------------------------

#define B_SZ 256
#define D_SZ 512
#define K_SZ 65536
#define TK   128   // memory rows per workgroup
#define DC   64    // D-chunk staged in LDS per iteration

typedef __bf16 v16bf __attribute__((ext_vector_type(16)));
typedef float  v8f   __attribute__((ext_vector_type(8)));
typedef unsigned int u32x4 __attribute__((ext_vector_type(4)));
typedef int          i32x8 __attribute__((ext_vector_type(8)));
typedef int          i32x4 __attribute__((ext_vector_type(4)));

#if defined(__has_builtin)
#if __has_builtin(__builtin_amdgcn_tensor_load_to_lds) && __has_builtin(__builtin_amdgcn_s_wait_tensorcnt)
#define HAVE_TDM 1
#endif
#endif
#ifndef HAVE_TDM
#define HAVE_TDM 0
#endif

// fp32 -> bf16 bits, round-to-nearest-even
__device__ inline unsigned bfbits(float x) {
  unsigned u = __float_as_uint(x);
  return (u + 0x7FFFu + ((u >> 16) & 1u)) >> 16;
}
__device__ inline unsigned pack_bf2(float x, float y) {
  return (bfbits(x) & 0xFFFFu) | (bfbits(y) << 16);
}
// monotone float<->int key for atomicMax on floats in LDS
__device__ inline int fkey(float v) { int b = __float_as_int(v); return b >= 0 ? b : (b ^ 0x7FFFFFFF); }
__device__ inline float fdec(int k) { return __int_as_float(k >= 0 ? k : (k ^ 0x7FFFFFFF)); }

// ---------------------------------------------------------------------------
// Kernel 1: normalize q1,q2; qm (fp32 + bf16), l_pos, pseudo labels.
// One block (256 thr) per batch row; 2 elements per thread.
// ---------------------------------------------------------------------------
__global__ __launch_bounds__(256) void prep_batch(
    const float* __restrict__ q1, const float* __restrict__ q2,
    const int* __restrict__ index, const int* __restrict__ label_table,
    float* __restrict__ qm_f32, unsigned short* __restrict__ qm_b,
    float* __restrict__ l_pos, int* __restrict__ pseudo)
{
  const int m = blockIdx.x, t = threadIdx.x;
  __shared__ float red0[256], red1[256], red2[256];
  const float* r1 = q1 + (size_t)m * D_SZ;
  const float* r2 = q2 + (size_t)m * D_SZ;
  const float a0 = r1[t], a1 = r1[t + 256];
  const float b0 = r2[t], b1 = r2[t + 256];
  red0[t] = a0 * a0 + a1 * a1;
  red1[t] = b0 * b0 + b1 * b1;
  red2[t] = a0 * b0 + a1 * b1;
  __syncthreads();
  for (int s = 128; s > 0; s >>= 1) {
    if (t < s) { red0[t] += red0[t + s]; red1[t] += red1[t + s]; red2[t] += red2[t + s]; }
    __syncthreads();
  }
  const float n1 = sqrtf(red0[0]) + 1e-12f;
  const float n2 = sqrtf(red1[0]) + 1e-12f;
  const float i1 = 1.f / n1, i2 = 1.f / n2;
  const float m0 = 0.5f * (a0 * i1 + b0 * i2);
  const float m1 = 0.5f * (a1 * i1 + b1 * i2);
  qm_f32[(size_t)m * D_SZ + t] = m0;
  qm_f32[(size_t)m * D_SZ + t + 256] = m1;
  qm_b[(size_t)m * D_SZ + t] = (unsigned short)bfbits(m0);
  qm_b[(size_t)m * D_SZ + t + 256] = (unsigned short)bfbits(m1);
  if (t == 0) {
    l_pos[m] = red2[0] * i1 * i2;
    pseudo[m] = label_table[index[m]];
  }
}

// ---------------------------------------------------------------------------
// Kernel 2: queue-entry labels + new_index_memory output.
// ---------------------------------------------------------------------------
__global__ __launch_bounds__(256) void prep_k(
    const int* __restrict__ index, const int* __restrict__ index_memory,
    const int* __restrict__ label_table, const int* __restrict__ curp,
    int* __restrict__ mem_lab, int* __restrict__ out_idx)
{
  const int i = blockIdx.x * 256 + threadIdx.x;
  const int im = index_memory[i];
  mem_lab[i] = (im == -1) ? -1 : label_table[im < 0 ? 0 : im];
  const int cur = *curp;
  const int d = (i - cur) & (K_SZ - 1);
  out_idx[i] = (d < B_SZ) ? index[d] : im;
}

// ---------------------------------------------------------------------------
// Kernel 3: fused GEMM (bf16 WMMA) + new_memory writeback + LSE partials.
// 512 workgroups x 256 threads (8 waves). Each WG: 128 memory rows vs 256
// batch rows. A (qm bf16) staged by the Tensor Data Mover; B (memory fp32)
// streamed, converted to bf16 for WMMA, and copied (with queue replacement)
// to d_out.
// ---------------------------------------------------------------------------
__global__ __launch_bounds__(256, 1) void moco_main(
    const float* __restrict__ memory, const float* __restrict__ qm_f32,
    const unsigned short* __restrict__ qm_b, const int* __restrict__ pseudo,
    const int* __restrict__ mem_lab, const int* __restrict__ curp,
    float* __restrict__ out, float* __restrict__ partials)
{
  __shared__ __align__(16) unsigned short sA[B_SZ][DC];  // qm tile   [m][k]
  __shared__ __align__(16) unsigned short sB[TK][DC];    // mem tile  [n][k]
  __shared__ int pseudo_s[B_SZ];
  __shared__ int labs_s[TK];
  __shared__ int maxp_i[B_SZ], maxn_i[B_SZ];
  __shared__ float sump_s[B_SZ], sumn_s[B_SZ];

  const int tid  = threadIdx.x;
  const int wv   = tid >> 5;
  const int lane = tid & 31;
  const int hb   = lane >> 4;
  const int l16  = lane & 15;
  const int wg   = blockIdx.x;
  const int kr0  = wg * TK;
  const int cur  = *curp;

  pseudo_s[tid] = pseudo[tid];
  if (tid < TK) labs_s[tid] = mem_lab[kr0 + tid];
  maxp_i[tid] = (int)0x80000000u;
  maxn_i[tid] = (int)0x80000000u;
  sump_s[tid] = 0.f;
  sumn_s[tid] = 0.f;
  if (cur == 0x7fffffff) sA[0][tid & (DC - 1)] = 0;  // never true: marks sA as written
  __syncthreads();

  v8f acc[16];
  const v8f z = {0, 0, 0, 0, 0, 0, 0, 0};
#pragma unroll
  for (int mt = 0; mt < 16; ++mt) acc[mt] = z;

  const int r_loc = tid >> 1;
  const int r = kr0 + r_loc;
  const int rep = (r - cur) & (K_SZ - 1);
  const bool replaced = rep < B_SZ;

  for (int dc = 0; dc < D_SZ / DC; ++dc) {
    // ---- stage A tile (qm bf16) via Tensor Data Mover ----
#if HAVE_TDM
    if (wv == 0) {
      const unsigned long long ga =
          (unsigned long long)(uintptr_t)qm_b + (unsigned long long)dc * DC * 2u;
      const unsigned lds = (unsigned)(uintptr_t)&sA[0][0];
      u32x4 g0 = { 1u,                                    // count=1
                   lds,                                   // lds_addr
                   (unsigned)(ga & 0xFFFFFFFFull),        // global_addr lo
                   (unsigned)((ga >> 32) & 0x1FFFFFFull) | (2u << 30) };  // hi | type=2
      i32x8 g1 = { (1 << 16),                                    // data_size=2B
                   (int)((D_SZ & 0xFFFF) << 16),                 // tensor_dim0 lo
                   (int)((D_SZ >> 16) | ((B_SZ & 0xFFFF) << 16)),// dim0 hi | dim1 lo
                   (int)((B_SZ >> 16) | (DC << 16)),             // dim1 hi | tile_dim0
                   B_SZ,                                         // tile_dim1
                   D_SZ,                                         // tensor_dim0_stride lo
                   0, 0 };
      i32x4 g2 = {0, 0, 0, 0}, g3 = {0, 0, 0, 0};
      i32x8 g4 = {0, 0, 0, 0, 0, 0, 0, 0};
      __builtin_amdgcn_tensor_load_to_lds(g0, g1, g2, g3, g4, 0);
    }
#else
    {
      const uint4* src = reinterpret_cast<const uint4*>(qm_b + (size_t)tid * D_SZ + dc * DC);
      uint4* dst = reinterpret_cast<uint4*>(&sA[tid][0]);
#pragma unroll
      for (int j = 0; j < (DC * 2) / 16; ++j) dst[j] = src[j];
    }
#endif
    // ---- stage B tile (fp32 -> bf16) and write new_memory output ----
    {
      const int c0 = dc * DC + (tid & 1) * 32;
      const int koff0 = (tid & 1) * 32;
#pragma unroll
      for (int j = 0; j < 8; ++j) {
        const int c = c0 + j * 4;
        float4 v = *reinterpret_cast<const float4*>(memory + (size_t)r * D_SZ + c);
        uint2 pk;
        pk.x = pack_bf2(v.x, v.y);
        pk.y = pack_bf2(v.z, v.w);
        *reinterpret_cast<uint2*>(&sB[r_loc][koff0 + j * 4]) = pk;
        float4 o = v;
        if (replaced) o = *reinterpret_cast<const float4*>(qm_f32 + (size_t)rep * D_SZ + c);
        const size_t ob = 1 + (size_t)r * D_SZ + c;  // d_out: [loss | new_memory | new_idx]
        out[ob + 0] = o.x; out[ob + 1] = o.y; out[ob + 2] = o.z; out[ob + 3] = o.w;
      }
    }
#if HAVE_TDM
    if (wv == 0) __builtin_amdgcn_s_wait_tensorcnt(0);
#endif
    __syncthreads();

    // ---- WMMA: wave wv owns 16-wide N-strip, all 16 M-tiles ----
#pragma unroll
    for (int ks = 0; ks < DC / 32; ++ks) {
      const int k0 = ks * 32 + hb * 8;
      v16bf bfrag;
      {
        const unsigned short* bp = &sB[wv * 16 + l16][0];
        reinterpret_cast<uint4*>(&bfrag)[0] = *reinterpret_cast<const uint4*>(bp + k0);
        reinterpret_cast<uint4*>(&bfrag)[1] = *reinterpret_cast<const uint4*>(bp + k0 + 16);
      }
#pragma unroll
      for (int mt = 0; mt < 16; ++mt) {
        v16bf afrag;
        const unsigned short* ap = &sA[mt * 16 + l16][0];
        reinterpret_cast<uint4*>(&afrag)[0] = *reinterpret_cast<const uint4*>(ap + k0);
        reinterpret_cast<uint4*>(&afrag)[1] = *reinterpret_cast<const uint4*>(ap + k0 + 16);
        acc[mt] = __builtin_amdgcn_wmma_f32_16x16x32_bf16(
            false, afrag, false, bfrag, (short)0, acc[mt], false, false);
      }
    }
    __syncthreads();
  }

  // ---- circle-loss logits -> per-WG streaming LSE partials ----
  const int nlab = labs_s[wv * 16 + l16];
  // pass 1: per-row maxima (atomicMax on monotone int key)
#pragma unroll
  for (int mt = 0; mt < 16; ++mt) {
#pragma unroll
    for (int j = 0; j < 8; ++j) {
      const int m = mt * 16 + (hb ? j + 8 : j);
      const float s = acc[mt][j];
      float vp, vn;
      if (pseudo_s[m] == nlab) {
        vp = -128.f * fmaxf(1.25f - s, 0.f) * (s - 0.75f);
        vn = -100007.f;  // gamma*relu(m)*(0-m) - BIG
      } else {
        vn = 128.f * fmaxf(s + 0.25f, 0.f) * (s - 0.25f);
        vp = -99879.f;   // -gamma*relu(1+m)*(0-(1-m)) - BIG
      }
      atomicMax(&maxp_i[m], fkey(vp));
      atomicMax(&maxn_i[m], fkey(vn));
    }
  }
  __syncthreads();
  // pass 2: sum of exp (ds_add_f32)
#pragma unroll
  for (int mt = 0; mt < 16; ++mt) {
#pragma unroll
    for (int j = 0; j < 8; ++j) {
      const int m = mt * 16 + (hb ? j + 8 : j);
      const float s = acc[mt][j];
      float vp, vn;
      if (pseudo_s[m] == nlab) {
        vp = -128.f * fmaxf(1.25f - s, 0.f) * (s - 0.75f);
        vn = -100007.f;
      } else {
        vn = 128.f * fmaxf(s + 0.25f, 0.f) * (s - 0.25f);
        vp = -99879.f;
      }
      atomicAdd(&sump_s[m], __expf(vp - fdec(maxp_i[m])));
      atomicAdd(&sumn_s[m], __expf(vn - fdec(maxn_i[m])));
    }
  }
  __syncthreads();
  float* p = partials + ((size_t)wg * B_SZ + tid) * 4;
  p[0] = fdec(maxp_i[tid]); p[1] = sump_s[tid];
  p[2] = fdec(maxn_i[tid]); p[3] = sumn_s[tid];
}

// ---------------------------------------------------------------------------
// Kernel 4: merge 512 partials per row, add l_pos column, softplus, mean/18.
// ---------------------------------------------------------------------------
__global__ __launch_bounds__(256) void moco_finalize(
    const float* __restrict__ partials, const float* __restrict__ l_pos,
    float* __restrict__ out)
{
  const int m = threadIdx.x;
  const float lp = l_pos[m];
  const float vp0 = -128.f * fmaxf(1.25f - lp, 0.f) * (lp - 0.75f);
  const float vn0 = -100007.f;
  float Mp = vp0, Mn = vn0;
  for (int g = 0; g < K_SZ / TK; ++g) {
    const float* p = partials + ((size_t)g * B_SZ + m) * 4;
    Mp = fmaxf(Mp, p[0]);
    Mn = fmaxf(Mn, p[2]);
  }
  float Sp = __expf(vp0 - Mp), Sn = __expf(vn0 - Mn);
  for (int g = 0; g < K_SZ / TK; ++g) {
    const float* p = partials + ((size_t)g * B_SZ + m) * 4;
    Sp += p[1] * __expf(p[0] - Mp);
    Sn += p[3] * __expf(p[2] - Mn);
  }
  const float x = (Mp + __logf(Sp)) + (Mn + __logf(Sn));
  const float sp = (x > 20.f) ? x : log1pf(__expf(x));
  __shared__ float red[256];
  red[m] = sp;
  __syncthreads();
  for (int s = 128; s > 0; s >>= 1) {
    if (m < s) red[m] += red[m + s];
    __syncthreads();
  }
  if (m == 0) out[0] = red[0] * (1.f / 256.f) / 18.f;
}

// ---------------------------------------------------------------------------
// Entry point. Inputs: q1, q2, memory, index, index_memory, label_table, cur.
// d_out (float): [loss(1) | new_memory(K*D) | new_index_memory(K) as int bits]
// Workspace layout (bytes):
//   0        qm_f32    256*512*4 = 524288
//   524288   qm_bf16   256*512*2 = 262144
//   786432   l_pos     256*4
//   787456   pseudo    256*4
//   788480   mem_lab   65536*4   = 262144
//   1050624  partials  512*256*4*4 = 2097152   (total ~3.2 MB)
// ---------------------------------------------------------------------------
extern "C" void kernel_launch(void* const* d_in, const int* in_sizes, int n_in,
                              void* d_out, int out_size, void* d_ws, size_t ws_size,
                              hipStream_t stream) {
  (void)in_sizes; (void)n_in; (void)out_size; (void)ws_size;
  const float* q1           = (const float*)d_in[0];
  const float* q2           = (const float*)d_in[1];
  const float* memory       = (const float*)d_in[2];
  const int*   index        = (const int*)d_in[3];
  const int*   index_memory = (const int*)d_in[4];
  const int*   label_table  = (const int*)d_in[5];
  const int*   cur          = (const int*)d_in[6];
  float* out = (float*)d_out;

  char* ws = (char*)d_ws;
  float*          qm_f32   = (float*)(ws + 0);
  unsigned short* qm_b     = (unsigned short*)(ws + 524288);
  float*          l_pos    = (float*)(ws + 786432);
  int*            pseudo   = (int*)(ws + 787456);
  int*            mem_lab  = (int*)(ws + 788480);
  float*          partials = (float*)(ws + 1050624);

  int* out_idx = (int*)d_out + 1 + (size_t)K_SZ * D_SZ;

  prep_batch<<<B_SZ, 256, 0, stream>>>(q1, q2, index, label_table,
                                       qm_f32, qm_b, l_pos, pseudo);
  prep_k<<<K_SZ / 256, 256, 0, stream>>>(index, index_memory, label_table, cur,
                                         mem_lab, out_idx);
  moco_main<<<K_SZ / TK, 256, 0, stream>>>(memory, qm_f32, qm_b, pseudo,
                                           mem_lab, cur, out, partials);
  moco_finalize<<<1, 256, 0, stream>>>(partials, l_pos, out);
}
